// LSTMSequence_4071628997380
// MI455X (gfx1250) — matrix operand order repaired and checked by reading, not compiled, not run on hardware
//
#include <hip/hip_runtime.h>
#include <hip/hip_bf16.h>

typedef __attribute__((ext_vector_type(16))) _Float16 v16h;
typedef __attribute__((ext_vector_type(8)))  _Float16 v8h;
typedef __attribute__((ext_vector_type(2)))  _Float16 v2h;
typedef __attribute__((ext_vector_type(8)))  float    v8f;
typedef __attribute__((ext_vector_type(4)))  float    v4f;

#define DD      8       // D*LEN_IN (input width)
#define HID     32
#define NGATE   128     // 4*HID
#define BB      256
#define TT      4096
#define OUTW    80      // D*LEN_OUT

#define ROWS    16      // batch rows per workgroup (one M-tile)
#define GSTRIDE 20      // f32 per gate column slot (16 rows + 4 pad) = 80B, conflict-free
#define HSTRIDE 40      // f16 per h row slot (32 K + 8 pad) = 80B, conflict-free, 16B-aligned

// ---- branchless activations on the gfx1250 trans pipe (v_tanh_f32) ----
__device__ __forceinline__ float fast_tanh(float x) {
#if __has_builtin(__builtin_amdgcn_tanhf)
    return __builtin_amdgcn_tanhf(x);
#else
    float a = __builtin_fabsf(x);
    float t = __expf(-2.0f * a);                       // t in (0,1], no overflow
    float r = (1.0f - t) * __builtin_amdgcn_rcpf(1.0f + t);
    return __builtin_copysignf(r, x);
#endif
}

__device__ __forceinline__ float sigm(float x) {
#if __has_builtin(__builtin_amdgcn_tanhf)
    return __builtin_fmaf(0.5f, __builtin_amdgcn_tanhf(0.5f * x), 0.5f);
#else
    return __builtin_amdgcn_rcpf(1.0f + __expf(-x));
#endif
}

// Build the 16x32 f16 A-operand from an LDS h-buffer stored [row][k] (HSTRIDE halves/row).
// Contiguous A-layout: lane m<16 holds K {0..15}; lane 16+m holds K {16..31}.
__device__ __forceinline__ v16h load_A_h(const _Float16* hbuf, int m, int hi) {
    const _Float16* p = hbuf + m * HSTRIDE;
    v8h lo = *(const v8h*)(p + 16 * hi);       // K = 16*hi .. 16*hi+7
    v8h up = *(const v8h*)(p + 16 * hi + 8);   // K = 16*hi+8 .. 16*hi+15
    return __builtin_shufflevector(lo, up, 0,1,2,3,4,5,6,7,8,9,10,11,12,13,14,15);
}

__global__ __launch_bounds__(256, 1)
void lstm_seq_kernel(const float* __restrict__ x,
                     const float* __restrict__ Wih1, const float* __restrict__ Whh1,
                     const float* __restrict__ bih1, const float* __restrict__ bhh1,
                     const float* __restrict__ Wih2, const float* __restrict__ Whh2,
                     const float* __restrict__ bih2, const float* __restrict__ bhh2,
                     const float* __restrict__ Wlin, const float* __restrict__ blin,
                     float* __restrict__ out)
{
    __shared__ __align__(16) float    g_gates[NGATE * GSTRIDE]; // [col][row] f32, padded stride
    __shared__ __align__(16) _Float16 g_h1[ROWS * HSTRIDE];     // [row][k] f16, padded stride
    __shared__ __align__(16) _Float16 g_h2[ROWS * HSTRIDE];

    const int tid    = threadIdx.x;
    const int lane   = tid & 31;
    const int waveId = tid >> 5;
    const int np     = lane & 15;    // row index m for A, column n' for B/C
    const int hi     = lane >> 4;
    const int bbase  = blockIdx.x * ROWS;
    const int ncol   = waveId * 16 + np;   // absolute gate/output column for this lane

    // wave-uniform scalar predicate for the output-projection waves (keeps EXEC
    // untouched around WMMA; s_cmp/s_cbranch instead of v_cmp/saveexec)
    const int waveIdS  = __builtin_amdgcn_readfirstlane(waveId);
    const bool outWave = (waveIdS < 5);

    // ---- Prologue: preload weight B-operands into registers (WMMA B layout) ----
    // B[k][n] = W[n][k]; lane (n', hi) holds halves h=0..15 -> k = 16*hi + h.
    v16h Bih1 = {};    // K padded: only k<8 nonzero (x width = 8)
    v16h Bhh1 = {}, Bih2 = {}, Bhh2 = {}, Blin = {};
    for (int h = 0; h < 16; ++h) {
        Bhh1[h] = (_Float16)Whh1[ncol * HID + 16 * hi + h];
        Bih2[h] = (_Float16)Wih2[ncol * HID + 16 * hi + h];
        Bhh2[h] = (_Float16)Whh2[ncol * HID + 16 * hi + h];
    }
    if (hi == 0) {
        for (int h = 0; h < DD; ++h) Bih1[h] = (_Float16)Wih1[ncol * DD + h];
    }
    if (outWave) {
        for (int h = 0; h < 16; ++h)
            Blin[h] = (_Float16)Wlin[ncol * HID + 16 * hi + h];
    }
    const float bias1 = bih1[ncol] + bhh1[ncol];
    const float bias2 = bih2[ncol] + bhh2[ncol];
    const float biasL = outWave ? blin[ncol] : 0.0f;

    // ---- zero initial hidden state; cell state lives in registers ----
    for (int i = tid; i < ROWS * HSTRIDE; i += 256) {
        g_h1[i] = (_Float16)0.0f;
        g_h2[i] = (_Float16)0.0f;
    }
    float c1a = 0.0f, c1b = 0.0f, c2a = 0.0f, c2b = 0.0f;

    // activation ownership: thread -> (row = tid&15, hidden pair 2*(tid>>4))
    const int arow = tid & 15;
    const int j0   = 2 * (tid >> 4);
    const int j1   = j0 + 1;

    // per-lane streaming pointers (advance by constant stride each step).
    // All 32 lanes load x for row np (valid addresses); lanes 16-31 zero it via
    // a precomputed 0/1 mask instead of a divergent branch.
    const float  xmask = (hi == 0) ? 1.0f : 0.0f;
    const float* xp = x + (size_t)(bbase + np) * TT * DD;
    float*       op = out + (size_t)(bbase + 8 * hi) * (size_t)TT * OUTW + ncol;  // waves 0-4 use it

    __syncthreads();

    for (int t = 0; t < TT; ++t) {
        // ======== layer 1: gates1 = x@Wih1^T + h1@Whh1^T + bias ========
        // Two INDEPENDENT WMMAs (no D->C chaining => no WMMA->WMMA RAW hazard
        // NOPs on the serial path); sum with packed f32 adds afterwards.
        v16h Ax = {};
        for (int k = 0; k < DD; ++k) Ax[k] = (_Float16)(xp[k] * xmask);  // K 0..7; K>=8 zero
        v16h Ah1 = load_A_h(g_h1, np, hi);
        v8f accA;
        for (int r = 0; r < 8; ++r) accA[r] = bias1;
        v8f accB = {};
        accA = __builtin_amdgcn_wmma_f32_16x16x32_f16(false, Ax,  false, Bih1, (short)0, accA, false, false);
        accB = __builtin_amdgcn_wmma_f32_16x16x32_f16(false, Ah1, false, Bhh1, (short)0, accB, false, false);
        v8f acc = accA + accB;
        {
            float* gp = &g_gates[ncol * GSTRIDE + 8 * hi];   // rows 8*hi..8*hi+7 contiguous
            v4f a0 = {acc[0], acc[1], acc[2], acc[3]};
            v4f a1 = {acc[4], acc[5], acc[6], acc[7]};
            *(v4f*)gp       = a0;
            *(v4f*)(gp + 4) = a1;
        }
        __syncthreads();   // B1: gates1 visible

        // ======== layer-1 activations (c1 in registers, h1 -> LDS f16) ========
        {
            float i0 = g_gates[(j0     ) * GSTRIDE + arow], i1 = g_gates[(j1     ) * GSTRIDE + arow];
            float f0 = g_gates[(32 + j0) * GSTRIDE + arow], f1 = g_gates[(32 + j1) * GSTRIDE + arow];
            float g0 = g_gates[(64 + j0) * GSTRIDE + arow], g1 = g_gates[(64 + j1) * GSTRIDE + arow];
            float o0 = g_gates[(96 + j0) * GSTRIDE + arow], o1 = g_gates[(96 + j1) * GSTRIDE + arow];
            c1a = sigm(f0) * c1a + sigm(i0) * fast_tanh(g0);
            c1b = sigm(f1) * c1b + sigm(i1) * fast_tanh(g1);
            float h0 = sigm(o0) * fast_tanh(c1a);
            float h1 = sigm(o1) * fast_tanh(c1b);
            v2h hp = {(_Float16)h0, (_Float16)h1};
            *(v2h*)&g_h1[arow * HSTRIDE + j0] = hp;
        }
        __syncthreads(); // B2: h1(t) visible

        // ======== layer 2: gates2 = h1@Wih2^T + h2@Whh2^T + bias ========
        v16h Ah1n = load_A_h(g_h1, np, hi);
        v16h Ah2  = load_A_h(g_h2, np, hi);
        v8f acc2A;
        for (int r = 0; r < 8; ++r) acc2A[r] = bias2;
        v8f acc2B = {};
        acc2A = __builtin_amdgcn_wmma_f32_16x16x32_f16(false, Ah1n, false, Bih2, (short)0, acc2A, false, false);
        acc2B = __builtin_amdgcn_wmma_f32_16x16x32_f16(false, Ah2,  false, Bhh2, (short)0, acc2B, false, false);
        v8f acc2 = acc2A + acc2B;
        {
            float* gp = &g_gates[ncol * GSTRIDE + 8 * hi];
            v4f a0 = {acc2[0], acc2[1], acc2[2], acc2[3]};
            v4f a1 = {acc2[4], acc2[5], acc2[6], acc2[7]};
            *(v4f*)gp       = a0;
            *(v4f*)(gp + 4) = a1;
        }
        __syncthreads();   // B3: gates2 visible

        // ======== layer-2 activations (c2 in registers, h2 -> LDS f16) ========
        {
            float i0 = g_gates[(j0     ) * GSTRIDE + arow], i1 = g_gates[(j1     ) * GSTRIDE + arow];
            float f0 = g_gates[(32 + j0) * GSTRIDE + arow], f1 = g_gates[(32 + j1) * GSTRIDE + arow];
            float g0 = g_gates[(64 + j0) * GSTRIDE + arow], g1 = g_gates[(64 + j1) * GSTRIDE + arow];
            float o0 = g_gates[(96 + j0) * GSTRIDE + arow], o1 = g_gates[(96 + j1) * GSTRIDE + arow];
            c2a = sigm(f0) * c2a + sigm(i0) * fast_tanh(g0);
            c2b = sigm(f1) * c2b + sigm(i1) * fast_tanh(g1);
            float h0 = sigm(o0) * fast_tanh(c2a);
            float h1 = sigm(o1) * fast_tanh(c2b);
            v2h hp = {(_Float16)h0, (_Float16)h1};
            *(v2h*)&g_h2[arow * HSTRIDE + j0] = hp;
        }
        __syncthreads();   // B4: h2(t) visible

        // ======== output projection: out(t) = h2@Wlin^T + b_lin (1 WMMA, waves 0-4) ========
        if (outWave) {     // scalar wave-uniform branch: EXEC all-ones inside -> WMMA legal
            v16h Ah2n = load_A_h(g_h2, np, hi);
            v8f accO;
            for (int r = 0; r < 8; ++r) accO[r] = biasL;
            accO = __builtin_amdgcn_wmma_f32_16x16x32_f16(false, Ah2n, false, Blin, (short)0, accO, false, false);
            for (int r = 0; r < 8; ++r)
                op[(size_t)r * TT * OUTW] = accO[r];   // row = 8*hi + r, col = ncol
        }

        xp += DD;      // next timestep of x (32B stride)
        op += OUTW;    // next timestep of out (320B stride)
    }
}

extern "C" void kernel_launch(void* const* d_in, const int* in_sizes, int n_in,
                              void* d_out, int out_size, void* d_ws, size_t ws_size,
                              hipStream_t stream) {
    (void)in_sizes; (void)n_in; (void)out_size; (void)d_ws; (void)ws_size;
    const float* x    = (const float*)d_in[0];
    const float* Wih1 = (const float*)d_in[1];
    const float* Whh1 = (const float*)d_in[2];
    const float* bih1 = (const float*)d_in[3];
    const float* bhh1 = (const float*)d_in[4];
    const float* Wih2 = (const float*)d_in[5];
    const float* Whh2 = (const float*)d_in[6];
    const float* bih2 = (const float*)d_in[7];
    const float* bhh2 = (const float*)d_in[8];
    const float* Wlin = (const float*)d_in[9];
    const float* blin = (const float*)d_in[10];

    lstm_seq_kernel<<<BB / ROWS, 256, 0, stream>>>(
        x, Wih1, Whh1, bih1, bhh1, Wih2, Whh2, bih2, bhh2, Wlin, blin,
        (float*)d_out);
}
